// MultiHeadAttention_53206054863428
// MI455X (gfx1250) — matrix operand order repaired and checked by reading, not compiled, not run on hardware
//
#include <hip/hip_runtime.h>
#include <hip/hip_bf16.h>

#define B_  4
#define S_  2048
#define D_  512
#define H_  8
#define HD_ 64

typedef __attribute__((ext_vector_type(16))) __bf16 v16bf;
typedef __attribute__((ext_vector_type(8)))  __bf16 v8bf;
typedef __attribute__((ext_vector_type(8)))  float  v8f;

// ---- fragment helpers (C/D per CDNA5 ISA 7.12.2; A/B use contiguous k, not table order) ----
// A 16x32 bf16: lane L holds row (L&15); e=0..15 -> K = ((L>>4)<<4)+e
// B 32x16 bf16: lane L holds col (L&15); e=0..15 -> K = ((L>>4)<<4)+e
// C/D 16x16 f32: vgpr r, lane L -> M = r + 8*(L>>4), N = (L&15)

static __device__ __forceinline__ v16bf cvt_a_frag(float4 x0, float4 x1,
                                                   float4 y0, float4 y1) {
  v16bf a;
  a[0] = (__bf16)x0.x; a[1] = (__bf16)x0.y; a[2]  = (__bf16)x0.z; a[3]  = (__bf16)x0.w;
  a[4] = (__bf16)x1.x; a[5] = (__bf16)x1.y; a[6]  = (__bf16)x1.z; a[7]  = (__bf16)x1.w;
  a[8] = (__bf16)y0.x; a[9] = (__bf16)y0.y; a[10] = (__bf16)y0.z; a[11] = (__bf16)y0.w;
  a[12] = (__bf16)y1.x; a[13] = (__bf16)y1.y; a[14] = (__bf16)y1.z; a[15] = (__bf16)y1.w;
  return a;
}

static __device__ __forceinline__ v16bf load_a_frag_f32(const float* rowk, int kA0) {
  return cvt_a_frag(*(const float4*)(rowk + kA0),
                    *(const float4*)(rowk + kA0 + 4),
                    *(const float4*)(rowk + kA0 + 8),
                    *(const float4*)(rowk + kA0 + 12));
}

static __device__ __forceinline__ v16bf make_a_frag_bf16(v8bf lo, v8bf hi) {
  return __builtin_shufflevector(lo, hi, 0,1,2,3,4,5,6,7,8,9,10,11,12,13,14,15);
}

static __device__ __forceinline__ v8f wmma_bf16(v16bf a, v16bf b, v8f c) {
  return __builtin_amdgcn_wmma_f32_16x16x32_bf16(false, a, false, b, (short)0, c,
                                                 false, false);
}

// ---- CDNA5 async global->LDS copy (CDNA5 ISA) ----
typedef __attribute__((address_space(3))) const void* lds_cptr_t;
static __device__ __forceinline__ unsigned lds_off_u32(const void* p) {
  return (unsigned)(uintptr_t)(lds_cptr_t)p;   // flat->local: wave LDS offset
}
static __device__ __forceinline__ void async_load_b128(unsigned lds_addr,
                                                       const void* gptr) {
  asm volatile("global_load_async_to_lds_b128 %0, %1, off"
               :: "v"(lds_addr), "v"((unsigned long long)(uintptr_t)gptr)
               : "memory");
}
static __device__ __forceinline__ void wait_async0() {
  asm volatile("s_wait_asynccnt 0x0" ::: "memory");
}

// ---- fp32 -> bf16 weight conversion ----
__global__ void __launch_bounds__(256)
mha_cvt_bf16_kernel(const float* __restrict__ src, __bf16* __restrict__ dst, int n) {
  int i = (blockIdx.x * 256 + threadIdx.x) * 4;
  if (i + 3 < n) {
    float4 v = *(const float4*)(src + i);
    dst[i + 0] = (__bf16)v.x; dst[i + 1] = (__bf16)v.y;
    dst[i + 2] = (__bf16)v.z; dst[i + 3] = (__bf16)v.w;
  }
}

// ---- QKV projection: Out = X @ W^T + bias; wave owns a 16x64 strip.
// W strip (64 rows x 512) staged once into LDS via async copy; A pipelined.
// mode 0: Out[b][h][s][hd]   (Q, K)      mode 1: Out[b][h][hd][s]  (V^T)
__global__ void __launch_bounds__(256)
mha_proj_kernel(const float* __restrict__ X, const __bf16* __restrict__ W,
                const float* __restrict__ bias, __bf16* __restrict__ Out, int mode) {
  __shared__ __bf16 Wlds[64 * D_];                 // 64 KB
  const int t    = threadIdx.x;
  const int lane = t & 31;
  const int wave = t >> 5;
  const int l15  = lane & 15;
  const int nb   = blockIdx.x * 4;                 // 4 N-tiles per wave
  const int tileM = blockIdx.y * 8 + wave;
  const int mrow  = tileM * 16 + l15;
  const int kA0   = (lane >> 4) << 4;
  const int kB0   = (lane >> 4) << 4;

  // stage W rows [nb*16, nb*16+64) -> LDS (contiguous 64 KB)
  {
    const char* wsrc = (const char*)(W + (size_t)nb * 16 * D_);
    const unsigned lbase = lds_off_u32(&Wlds[0]);
    #pragma unroll
    for (int c = 0; c < 16; ++c) {
      const unsigned off = (unsigned)((c * 256 + t) * 16);
      async_load_b128(lbase + off, wsrc + off);
    }
    wait_async0();
  }
  __syncthreads();

  const float* xrow = X + (size_t)mrow * D_;
  const __bf16* wl0 = &Wlds[(size_t)(0 * 16 + l15) * D_];
  const __bf16* wl1 = &Wlds[(size_t)(1 * 16 + l15) * D_];
  const __bf16* wl2 = &Wlds[(size_t)(2 * 16 + l15) * D_];
  const __bf16* wl3 = &Wlds[(size_t)(3 * 16 + l15) * D_];

  // software pipeline: raw fp32 A data for step k lives while k+32 loads fly
  float4 rx0 = *(const float4*)(xrow + kA0);
  float4 rx1 = *(const float4*)(xrow + kA0 + 4);
  float4 ry0 = *(const float4*)(xrow + kA0 + 8);
  float4 ry1 = *(const float4*)(xrow + kA0 + 12);
  v8f acc0 = {}, acc1 = {}, acc2 = {}, acc3 = {};
  #pragma unroll 4
  for (int k0 = 0; k0 < D_; k0 += 32) {
    v16bf a = cvt_a_frag(rx0, rx1, ry0, ry1);
    const int kn = (k0 + 32) & (D_ - 1);           // wrap: no branch near WMMA
    rx0 = *(const float4*)(xrow + kn + kA0);
    rx1 = *(const float4*)(xrow + kn + kA0 + 4);
    ry0 = *(const float4*)(xrow + kn + kA0 + 8);
    ry1 = *(const float4*)(xrow + kn + kA0 + 12);
    __builtin_prefetch(xrow + kn + 128, 0, 3);
    v16bf b0 = *(const v16bf*)(wl0 + k0 + kB0);
    v16bf b1 = *(const v16bf*)(wl1 + k0 + kB0);
    v16bf b2 = *(const v16bf*)(wl2 + k0 + kB0);
    v16bf b3 = *(const v16bf*)(wl3 + k0 + kB0);
    acc0 = wmma_bf16(a, b0, acc0);
    acc1 = wmma_bf16(a, b1, acc1);
    acc2 = wmma_bf16(a, b2, acc2);
    acc3 = wmma_bf16(a, b3, acc3);
  }

#define PROJ_STORE(ACC, NT)                                                    \
  {                                                                            \
    const int ncol = (nb + (NT)) * 16 + l15;                                   \
    const float bval = bias[ncol];                                             \
    const int h = ncol >> 6, hd = ncol & 63;                                   \
    _Pragma("unroll")                                                          \
    for (int r = 0; r < 8; ++r) {                                              \
      const int mm = tileM * 16 + r + ((lane >> 4) << 3);                      \
      const int bb = mm >> 11, ss = mm & 2047;                                 \
      const float v = (ACC)[r] + bval;                                         \
      size_t off;                                                              \
      if (mode == 0) off = (((size_t)(bb * H_ + h)) * S_ + ss) * HD_ + hd;     \
      else           off = (((size_t)(bb * H_ + h)) * HD_ + hd) * S_ + ss;     \
      Out[off] = (__bf16)v;                                                    \
    }                                                                          \
  }
  PROJ_STORE(acc0, 0)
  PROJ_STORE(acc1, 1)
  PROJ_STORE(acc2, 2)
  PROJ_STORE(acc3, 3)
#undef PROJ_STORE
}

// ---- fused attention: scores -> softmax -> (write attn) -> probs @ V ----
// One workgroup per (bh, 16-row block). Scores tile lives in LDS (128 KB).
__global__ void __launch_bounds__(256)
mha_attn_fused_kernel(const __bf16* __restrict__ Qh, const __bf16* __restrict__ Kh,
                      const __bf16* __restrict__ Vt, const int* __restrict__ mask,
                      float* __restrict__ attn, __bf16* __restrict__ Ctx) {
  __shared__ float sc[16 * S_];        // 128 KB: scores, then probabilities
  __shared__ float pred[4 * 16 * 16];  // 4 KB: PV partial tiles
  __shared__ float rbuf[256];
  __shared__ float rstat[16];

  const int t    = threadIdx.x;
  const int lane = t & 31;
  const int w    = t >> 5;
  const int l15  = lane & 15;
  const int rb   = blockIdx.x;         // 0..127 row block
  const int bh   = blockIdx.y;         // 0..31
  const int bb   = bh >> 3, hh = bh & 7;
  const int kA0  = (lane >> 4) << 4;
  const int kB0  = (lane >> 4) << 4;
  const int mrow = rb * 16 + l15;

  // phase 1: scores (Q . K^T)/8 + mask -> LDS (B-frags pipelined)
  const __bf16* qrow = Qh + ((size_t)bh * S_ + mrow) * HD_;
  v16bf qa0 = make_a_frag_bf16(*(const v8bf*)(qrow + kA0),
                               *(const v8bf*)(qrow + kA0 + 8));
  v16bf qa1 = make_a_frag_bf16(*(const v8bf*)(qrow + 32 + kA0),
                               *(const v8bf*)(qrow + 32 + kA0 + 8));
  const __bf16* kbase = Kh + ((size_t)bh * S_ + w * 256 + l15) * HD_;
  v16bf kb0 = *(const v16bf*)(kbase + kB0);
  v16bf kb1 = *(const v16bf*)(kbase + 32 + kB0);
  for (int nt = 0; nt < 16; ++nt) {
    const __bf16* krown = kbase + (size_t)(((nt + 1) & 15) * 16) * HD_;
    v16bf nb0 = *(const v16bf*)(krown + kB0);
    v16bf nb1 = *(const v16bf*)(krown + 32 + kB0);
    v8f acc = {};
    acc = wmma_bf16(qa0, kb0, acc);
    acc = wmma_bf16(qa1, kb1, acc);
    const int ncol = w * 256 + nt * 16 + l15;
    #pragma unroll
    for (int r = 0; r < 8; ++r) {
      const int mrel = r + ((lane >> 4) << 3);
      const int mv = mask[((size_t)bb * S_ + rb * 16 + mrel) * S_ + ncol];
      float v = acc[r] * 0.125f;
      if (mv == 0) v = -1e20f;
      sc[mrel * S_ + ncol] = v;
    }
    kb0 = nb0; kb1 = nb1;
  }
  __syncthreads();

  // phase 2: softmax over each of the 16 rows (16 threads per row)
  const int row = t >> 4;
  const int seg = t & 15;
  float* rp = sc + row * S_ + seg * 128;
  float mx = -3.0e38f;
  #pragma unroll 8
  for (int i = 0; i < 32; ++i) {
    float4 v = ((float4*)rp)[i];
    mx = fmaxf(mx, fmaxf(fmaxf(v.x, v.y), fmaxf(v.z, v.w)));
  }
  rbuf[t] = mx;
  __syncthreads();
  if (t < 16) {
    float m = rbuf[t * 16];
    #pragma unroll
    for (int i = 1; i < 16; ++i) m = fmaxf(m, rbuf[t * 16 + i]);
    rstat[t] = m;
  }
  __syncthreads();
  mx = rstat[row];
  float sum = 0.f;
  #pragma unroll 4
  for (int i = 0; i < 32; ++i) {
    float4 v = ((float4*)rp)[i];
    v.x = __expf(v.x - mx); v.y = __expf(v.y - mx);
    v.z = __expf(v.z - mx); v.w = __expf(v.w - mx);
    sum += (v.x + v.y) + (v.z + v.w);
    ((float4*)rp)[i] = v;
  }
  rbuf[t] = sum;
  __syncthreads();
  if (t < 16) {
    float s = 0.f;
    #pragma unroll
    for (int i = 0; i < 16; ++i) s += rbuf[t * 16 + i];
    rstat[t] = 1.0f / s;
  }
  __syncthreads();
  const float inv = rstat[row];
  float* gp = attn + ((size_t)bh * S_ + rb * 16 + row) * S_ + seg * 128;
  #pragma unroll 4
  for (int i = 0; i < 32; ++i) {
    float4 v = ((float4*)rp)[i];
    v.x *= inv; v.y *= inv; v.z *= inv; v.w *= inv;
    ((float4*)rp)[i] = v;   // probs stay in LDS for phase 3
    ((float4*)gp)[i] = v;   // attention output: single HBM write
  }
  __syncthreads();

  // phase 3: ctx = probs @ V; waves split K in half, 4 N-tiles (HD=64)
  const int ntile = w & 3;
  const int khalf = w >> 2;
  const int ncol  = ntile * 16 + l15;
  const __bf16* vrow = Vt + ((size_t)bh * HD_ + ncol) * S_ + khalf * 1024;
  const float*  prow = sc + l15 * S_ + khalf * 1024;
  v8f acc = {};
  v16bf bcur = *(const v16bf*)(vrow + kB0);
  for (int k0 = 0; k0 < 1024; k0 += 32) {
    const int kn = (k0 + 32) & 1023;
    v16bf bnext = *(const v16bf*)(vrow + kn + kB0);
    v16bf a = load_a_frag_f32(prow + k0, kA0);   // ds_load + cvt
    acc = wmma_bf16(a, bcur, acc);
    bcur = bnext;
  }
  if (w >= 4) {
    #pragma unroll
    for (int r = 0; r < 8; ++r)
      pred[ntile * 256 + (r + ((lane >> 4) << 3)) * 16 + l15] = acc[r];
  }
  __syncthreads();
  if (w < 4) {
    #pragma unroll
    for (int r = 0; r < 8; ++r) {
      const int mrel = r + ((lane >> 4) << 3);
      const float v = acc[r] + pred[ntile * 256 + mrel * 16 + l15];
      Ctx[(((size_t)bb * S_ + rb * 16 + mrel) * H_ + hh) * HD_ + ncol] = (__bf16)v;
    }
  }
}

// ---- output projection: out = ctx @ Wo^T + bo (fp32), 16x64 per wave ----
__global__ void __launch_bounds__(256)
mha_outproj_kernel(const __bf16* __restrict__ Ctx, const __bf16* __restrict__ W,
                   const float* __restrict__ bias, float* __restrict__ Out) {
  __shared__ __bf16 Wlds[64 * D_];                 // 64 KB
  const int t    = threadIdx.x;
  const int lane = t & 31;
  const int wave = t >> 5;
  const int l15  = lane & 15;
  const int nb   = blockIdx.x * 4;
  const int tileM = blockIdx.y * 8 + wave;
  const int mrow  = tileM * 16 + l15;
  const int kA0   = (lane >> 4) << 4;
  const int kB0   = (lane >> 4) << 4;

  {
    const char* wsrc = (const char*)(W + (size_t)nb * 16 * D_);
    const unsigned lbase = lds_off_u32(&Wlds[0]);
    #pragma unroll
    for (int c = 0; c < 16; ++c) {
      const unsigned off = (unsigned)((c * 256 + t) * 16);
      async_load_b128(lbase + off, wsrc + off);
    }
    wait_async0();
  }
  __syncthreads();

  const __bf16* crow = Ctx + (size_t)mrow * D_;
  const __bf16* wl0 = &Wlds[(size_t)(0 * 16 + l15) * D_];
  const __bf16* wl1 = &Wlds[(size_t)(1 * 16 + l15) * D_];
  const __bf16* wl2 = &Wlds[(size_t)(2 * 16 + l15) * D_];
  const __bf16* wl3 = &Wlds[(size_t)(3 * 16 + l15) * D_];

  v8bf alo = *(const v8bf*)(crow + kA0);
  v8bf ahi = *(const v8bf*)(crow + kA0 + 8);
  v8f acc0 = {}, acc1 = {}, acc2 = {}, acc3 = {};
  #pragma unroll 4
  for (int k0 = 0; k0 < D_; k0 += 32) {
    v16bf a = make_a_frag_bf16(alo, ahi);
    const int kn = (k0 + 32) & (D_ - 1);
    alo = *(const v8bf*)(crow + kn + kA0);
    ahi = *(const v8bf*)(crow + kn + kA0 + 8);
    v16bf b0 = *(const v16bf*)(wl0 + k0 + kB0);
    v16bf b1 = *(const v16bf*)(wl1 + k0 + kB0);
    v16bf b2 = *(const v16bf*)(wl2 + k0 + kB0);
    v16bf b3 = *(const v16bf*)(wl3 + k0 + kB0);
    acc0 = wmma_bf16(a, b0, acc0);
    acc1 = wmma_bf16(a, b1, acc1);
    acc2 = wmma_bf16(a, b2, acc2);
    acc3 = wmma_bf16(a, b3, acc3);
  }

#define OUT_STORE(ACC, NT)                                                     \
  {                                                                            \
    const int ncol = (nb + (NT)) * 16 + l15;                                   \
    const float bval = bias[ncol];                                             \
    _Pragma("unroll")                                                          \
    for (int r = 0; r < 8; ++r) {                                              \
      const int mm = tileM * 16 + r + ((lane >> 4) << 3);                      \
      Out[(size_t)mm * D_ + ncol] = (ACC)[r] + bval;                           \
    }                                                                          \
  }
  OUT_STORE(acc0, 0)
  OUT_STORE(acc1, 1)
  OUT_STORE(acc2, 2)
  OUT_STORE(acc3, 3)
#undef OUT_STORE
}

extern "C" void kernel_launch(void* const* d_in, const int* in_sizes, int n_in,
                              void* d_out, int out_size, void* d_ws, size_t ws_size,
                              hipStream_t stream) {
  (void)in_sizes; (void)n_in; (void)out_size; (void)ws_size;
  const float* query = (const float*)d_in[0];
  const float* key_  = (const float*)d_in[1];
  const float* value = (const float*)d_in[2];
  const int*   mask  = (const int*)d_in[3];
  const float* Wq = (const float*)d_in[4];
  const float* bq = (const float*)d_in[5];
  const float* Wk = (const float*)d_in[6];
  const float* bk = (const float*)d_in[7];
  const float* Wv = (const float*)d_in[8];
  const float* bv = (const float*)d_in[9];
  const float* Wo = (const float*)d_in[10];
  const float* bo = (const float*)d_in[11];

  float* out  = (float*)d_out;                         // [B,S,D] fp32
  float* attn = out + (size_t)B_ * S_ * D_;            // [B,H,S,S] fp32

  char* ws = (char*)d_ws;
  const size_t WB = (size_t)D_ * D_ * sizeof(__bf16);            // 512 KB
  const size_t AB = (size_t)B_ * H_ * S_ * HD_ * sizeof(__bf16); // 8 MB
  __bf16* Wq_bf = (__bf16*)(ws + 0 * WB);
  __bf16* Wk_bf = (__bf16*)(ws + 1 * WB);
  __bf16* Wv_bf = (__bf16*)(ws + 2 * WB);
  __bf16* Wo_bf = (__bf16*)(ws + 3 * WB);
  __bf16* Qh  = (__bf16*)(ws + 4 * WB);
  __bf16* Kh  = (__bf16*)(ws + 4 * WB + 1 * AB);
  __bf16* Vt  = (__bf16*)(ws + 4 * WB + 2 * AB);
  __bf16* Ctx = (__bf16*)(ws + 4 * WB + 3 * AB);

  dim3 blk(256);

  // 1) weights -> bf16
  mha_cvt_bf16_kernel<<<D_ * D_ / 1024, blk, 0, stream>>>(Wq, Wq_bf, D_ * D_);
  mha_cvt_bf16_kernel<<<D_ * D_ / 1024, blk, 0, stream>>>(Wk, Wk_bf, D_ * D_);
  mha_cvt_bf16_kernel<<<D_ * D_ / 1024, blk, 0, stream>>>(Wv, Wv_bf, D_ * D_);
  mha_cvt_bf16_kernel<<<D_ * D_ / 1024, blk, 0, stream>>>(Wo, Wo_bf, D_ * D_);

  // 2) Q/K/V projections (8 waves/block, 16x64 strip per wave, W via LDS)
  dim3 gproj(D_ / 64, (B_ * S_) / (16 * 8));           // (8, 64)
  mha_proj_kernel<<<gproj, blk, 0, stream>>>(query, Wq_bf, bq, Qh, 0);
  mha_proj_kernel<<<gproj, blk, 0, stream>>>(key_,  Wk_bf, bk, Kh, 0);
  mha_proj_kernel<<<gproj, blk, 0, stream>>>(value, Wv_bf, bv, Vt, 1);

  // 3) fused scores + softmax + attn-write + PV
  dim3 gfa(S_ / 16, B_ * H_);                          // (128, 32)
  mha_attn_fused_kernel<<<gfa, blk, 0, stream>>>(Qh, Kh, Vt, mask, attn, Ctx);

  // 4) output projection
  dim3 gop(D_ / 64, (B_ * S_) / (16 * 8));             // (8, 64)
  mha_outproj_kernel<<<gop, blk, 0, stream>>>(Ctx, Wo_bf, bo, out);
}
